// PNA_19997367730369
// MI455X (gfx1250) — matrix operand order repaired and checked by reading, not compiled, not run on hardware
//
#include <hip/hip_runtime.h>
#include <math.h>

// ---------------- types for WMMA (CDNA5 / gfx1250, wave32) ----------------
typedef __attribute__((ext_vector_type(16))) __bf16 v16bf;
typedef __attribute__((ext_vector_type(8)))  float  v8f;

__device__ __forceinline__ unsigned short f2bf(float f) {
    unsigned u = __float_as_uint(f);
    unsigned r = u + 0x7fffu + ((u >> 16) & 1u);   // RNE (used off critical path)
    return (unsigned short)(r >> 16);
}

// pack two floats -> two bf16 in one dword: per-element round-half-up + v_perm_b32
__device__ __forceinline__ unsigned pack_bf2(float lo, float hi) {
    return __builtin_amdgcn_perm(__float_as_uint(hi) + 0x8000u,
                                 __float_as_uint(lo) + 0x8000u, 0x07060302u);
}

#define AVG_LOG 3.4965075614664802f
#define BN_EPS 1e-5f
#define STD_EPS 1e-5f
#define NGRAPH 64
#define FOUT_LAST 20
#define NCLASS 11

// ---------------- graph preprocessing ----------------
__global__ void degree_kernel(const int* __restrict__ dst, int* __restrict__ cnt, int E) {
    int i = blockIdx.x * blockDim.x + threadIdx.x;
    if (i < E) atomicAdd(&cnt[dst[i]], 1);
}

__global__ __launch_bounds__(1024) void scan_kernel(const int* __restrict__ cnt,
                                                    int* __restrict__ rowptr, int n) {
    __shared__ int buf[1024];
    __shared__ int carry;
    int tid = threadIdx.x;
    if (tid == 0) { carry = 0; rowptr[0] = 0; }
    __syncthreads();
    for (int base = 0; base < n; base += 1024) {
        int i = base + tid;
        int v = (i < n) ? cnt[i] : 0;
        buf[tid] = v;
        __syncthreads();
        for (int off = 1; off < 1024; off <<= 1) {
            int t = (tid >= off) ? buf[tid - off] : 0;
            __syncthreads();
            buf[tid] += t;
            __syncthreads();
        }
        if (i < n) rowptr[i + 1] = buf[tid] + carry;
        __syncthreads();
        if (tid == 0) carry += buf[1023];
        __syncthreads();
    }
}

__global__ void scatter_kernel(const int* __restrict__ src, const int* __restrict__ dst,
                               const int* __restrict__ rowptr, int* __restrict__ cursor,
                               int* __restrict__ csr, int E) {
    int i = blockIdx.x * blockDim.x + threadIdx.x;
    if (i >= E) return;
    int d = dst[i];
    int pos = rowptr[d] + atomicAdd(&cursor[d], 1);
    csr[pos] = src[i];
}

__global__ void node_scalars_kernel(const int* __restrict__ cnt, float* __restrict__ invdeg,
                                    float* __restrict__ s1, float* __restrict__ s2, int N) {
    int i = blockIdx.x * blockDim.x + threadIdx.x;
    if (i >= N) return;
    float degc = fmaxf((float)cnt[i], 1.0f);
    float logd = logf(degc + 1.0f);
    invdeg[i] = 1.0f / degc;
    s1[i] = logd * (1.0f / AVG_LOG);
    s2[i] = AVG_LOG / logd;
}

// ---------------- per-node aggregation (one wave per node, atomic-free) ----------------
template<int NF>
__global__ __launch_bounds__(128) void agg_kernel(const float* __restrict__ Hin,
                                                  const int* __restrict__ rowptr,
                                                  const int* __restrict__ csr,
                                                  const float* __restrict__ invdeg,
                                                  float* __restrict__ AGG, int N) {
    const int F = NF * 32;
    int node = blockIdx.x * 4 + (threadIdx.x >> 5);
    if (node >= N) return;
    int lane = threadIdx.x & 31;
    int beg = rowptr[node], end = rowptr[node + 1];
    float sum[NF], sq[NF], mn[NF], mx[NF];
#pragma unroll
    for (int t = 0; t < NF; ++t) { sum[t] = 0.f; sq[t] = 0.f; mn[t] = 3.4e38f; mx[t] = -3.4e38f; }
    for (int e = beg; e < end; ++e) {
        const float* rowp = Hin + (size_t)csr[e] * F;
        if (e + 1 < end) __builtin_prefetch(Hin + (size_t)csr[e + 1] * F, 0, 3);
#pragma unroll
        for (int t = 0; t < NF; ++t) {
            float v = rowp[t * 32 + lane];
            sum[t] += v; sq[t] += v * v;
            mn[t] = fminf(mn[t], v); mx[t] = fmaxf(mx[t], v);
        }
    }
    float inv = invdeg[node];
    bool has = end > beg;
    float* out = AGG + (size_t)node * 4 * F;
#pragma unroll
    for (int t = 0; t < NF; ++t) {
        int c = t * 32 + lane;
        float mean = sum[t] * inv;
        float var = sq[t] * inv - mean * mean;
        out[c]         = mean;
        out[F + c]     = has ? mn[t] : 0.f;
        out[2 * F + c] = has ? mx[t] : 0.f;
        out[3 * F + c] = sqrtf(fmaxf(var, 0.f) + STD_EPS);
    }
}

// ---------------- weight prep: transpose to [Np, 13F] bf16, zero-pad cols ----------------
__global__ void wprep_kernel(const float* __restrict__ W, unsigned short* __restrict__ Wt,
                             int K13, int Fo, int total) {
    int i = blockIdx.x * blockDim.x + threadIdx.x;
    if (i >= total) return;
    int n = i / K13, k = i - n * K13;
    Wt[i] = (n < Fo) ? f2bf(W[(size_t)k * Fo + n]) : (unsigned short)0;
}

// ---------------- fused combine + GEMM via v_wmma_f32_16x16x32_bf16 ----------------
// Virtual H13 row: [ x(F) | agg(4F) | s1*agg(4F) | s2*agg(4F) ] ; all boundaries %32==0.
// One wave computes a 16-row M tile against ALL N tiles (acc[NTN]) so the A-fragment
// build (gather + cvt) is amortized over NTN WMMAs per K-chunk.
template<int F, int Fo>
__global__ __launch_bounds__(128) void pna_gemm_kernel(
    const float* __restrict__ Hin, const float* __restrict__ AGG,
    const float* __restrict__ s1a, const float* __restrict__ s2a,
    const unsigned short* __restrict__ Wt, const float* __restrict__ bias,
    float* __restrict__ Out, int mtiles) {
    constexpr int K13 = 13 * F;
    constexpr int F4  = 4 * F;
    constexpr int NTN = (Fo + 15) / 16;

    int mt = blockIdx.x * 4 + (threadIdx.x >> 5);
    if (mt >= mtiles) return;                 // wave-uniform: EXEC stays all-ones
    int lane = threadIdx.x & 31;
    int g = lane >> 4;                        // lane half selects K-stripe
    int m = lane & 15;                        // A row / B,C column index
    int arow = mt * 16 + m;
    float s1v = s1a[arow], s2v = s2a[arow];
    const float* xrow = Hin + (size_t)arow * F;
    const float* grow = AGG + (size_t)arow * F4;

    size_t wb[NTN];
#pragma unroll
    for (int nt = 0; nt < NTN; ++nt) wb[nt] = (size_t)(nt * 16 + m) * K13;

    v8f acc[NTN];
    v8f vzero = {0.f, 0.f, 0.f, 0.f, 0.f, 0.f, 0.f, 0.f};
#pragma unroll
    for (int nt = 0; nt < NTN; ++nt) acc[nt] = vzero;

    auto chunk = [&](const float* p, float sc, int Kg) {
        // A fragment (16x32 bf16): lane half g covers K = Kg+g*16..+15
        const float* q0 = p + g * 16;
        const float* q1 = p + g * 16 + 8;
        float4 x0 = *reinterpret_cast<const float4*>(q0);
        float4 x1 = *reinterpret_cast<const float4*>(q0 + 4);
        float4 y0 = *reinterpret_cast<const float4*>(q1);
        float4 y1 = *reinterpret_cast<const float4*>(q1 + 4);
        union { unsigned d[8]; v16bf v; } av;
        av.d[0] = pack_bf2(x0.x * sc, x0.y * sc);
        av.d[1] = pack_bf2(x0.z * sc, x0.w * sc);
        av.d[2] = pack_bf2(x1.x * sc, x1.y * sc);
        av.d[3] = pack_bf2(x1.z * sc, x1.w * sc);
        av.d[4] = pack_bf2(y0.x * sc, y0.y * sc);
        av.d[5] = pack_bf2(y0.z * sc, y0.w * sc);
        av.d[6] = pack_bf2(y1.x * sc, y1.y * sc);
        av.d[7] = pack_bf2(y1.z * sc, y1.w * sc);
#pragma unroll
        for (int nt = 0; nt < NTN; ++nt) {
            // B fragment (32x16 bf16): lane n=m holds K = Kg + g*16 + j, contiguous in Wt
            v16bf bv = *reinterpret_cast<const v16bf*>(Wt + wb[nt] + Kg + g * 16);
            acc[nt] = __builtin_amdgcn_wmma_f32_16x16x32_bf16(false, av.v, false, bv,
                                                              (short)0, acc[nt], false, false);
        }
    };

    // self features: K in [0, F)
    for (int kb = 0; kb < F; kb += 32) chunk(xrow + kb, 1.0f, kb);
    // aggregated features, 3 scaler passes over the same AGG columns
#pragma unroll
    for (int s = 0; s < 3; ++s) {
        float sc = (s == 0) ? 1.0f : ((s == 1) ? s1v : s2v);
        int base = F + s * F4;
        for (int kb = 0; kb < F4; kb += 32) chunk(grow + kb, sc, base + kb);
    }

#pragma unroll
    for (int nt = 0; nt < NTN; ++nt) {
        int col = nt * 16 + m;
        bool ok = ((Fo & 15) == 0) || (col < Fo);
        float bcol = ok ? bias[col] : 0.f;
#pragma unroll
        for (int r = 0; r < 8; ++r) {
            int orow = mt * 16 + r + 8 * g;   // C/D: VGPR r -> M=r (lanes 0-15), M=r+8 (16-31)
            if (ok) Out[(size_t)orow * Fo + col] = acc[nt][r] + bcol;
        }
    }
}

// ---------------- BatchNorm + ReLU ----------------
__global__ __launch_bounds__(256) void bn_stats_kernel(const float* __restrict__ X,
                                                       float* __restrict__ gs,
                                                       int total, int Fo) {
    __shared__ float ls[192];
    for (int i = threadIdx.x; i < 2 * Fo; i += blockDim.x) ls[i] = 0.f;
    __syncthreads();
    int stride = gridDim.x * blockDim.x;
    for (int i = blockIdx.x * blockDim.x + threadIdx.x; i < total; i += stride) {
        float v = X[i];
        int c = i % Fo;
        atomicAdd(&ls[c], v);
        atomicAdd(&ls[Fo + c], v * v);
    }
    __syncthreads();
    for (int i = threadIdx.x; i < 2 * Fo; i += blockDim.x) atomicAdd(&gs[i], ls[i]);
}

__global__ void bn_apply_kernel(float* __restrict__ X, const float* __restrict__ gs,
                                const float* __restrict__ gamma, const float* __restrict__ beta,
                                int total, int Fo, float invN, int use_bn, int use_relu) {
    int i = blockIdx.x * blockDim.x + threadIdx.x;
    if (i >= total) return;
    float v = X[i];
    if (use_bn) {
        int c = i % Fo;
        float mu = gs[c] * invN;
        float var = gs[Fo + c] * invN - mu * mu;
        v = (v - mu) * rsqrtf(var + BN_EPS) * gamma[c] + beta[c];
    }
    if (use_relu) v = fmaxf(v, 0.f);
    X[i] = v;
}

// ---------------- global mean pool + head ----------------
__global__ void pool_cnt_kernel(const int* __restrict__ batch, float* __restrict__ gcnt, int N) {
    int i = blockIdx.x * blockDim.x + threadIdx.x;
    if (i < N) atomicAdd(&gcnt[batch[i]], 1.0f);
}

__global__ void pool_sum_kernel(const float* __restrict__ H, const int* __restrict__ batch,
                                float* __restrict__ zsum, int N) {
    int idx = blockIdx.x * blockDim.x + threadIdx.x;
    if (idx >= N * FOUT_LAST) return;
    int n = idx / FOUT_LAST;
    int c = idx - n * FOUT_LAST;
    atomicAdd(&zsum[batch[n] * FOUT_LAST + c], H[idx]);
}

__global__ __launch_bounds__(64) void head_kernel(const float* __restrict__ zsum,
                                                  const float* __restrict__ gcnt,
                                                  const float* __restrict__ wl,
                                                  const float* __restrict__ bl,
                                                  float* __restrict__ out) {
    int gi = threadIdx.x;
    if (gi >= NGRAPH) return;
    float z[FOUT_LAST];
    float ic = 1.0f / fmaxf(gcnt[gi], 1.0f);
#pragma unroll
    for (int c = 0; c < FOUT_LAST; ++c) {
        z[c] = zsum[gi * FOUT_LAST + c] * ic;
        out[NGRAPH * NCLASS + gi * FOUT_LAST + c] = z[c];   // second output: z
    }
    float l[NCLASS];
    float mx = -3.4e38f;
#pragma unroll
    for (int j = 0; j < NCLASS; ++j) {
        float a = bl[j];
#pragma unroll
        for (int c = 0; c < FOUT_LAST; ++c) a += z[c] * wl[c * NCLASS + j];
        l[j] = a;
        mx = fmaxf(mx, a);
    }
    float s = 0.f;
#pragma unroll
    for (int j = 0; j < NCLASS; ++j) { l[j] = expf(l[j] - mx); s += l[j]; }
    float is = 1.0f / s;
#pragma unroll
    for (int j = 0; j < NCLASS; ++j) out[gi * NCLASS + j] = l[j] * is;   // first output: softmax
}

// ---------------- host orchestration ----------------
extern "C" void kernel_launch(void* const* d_in, const int* in_sizes, int n_in,
                              void* d_out, int out_size, void* d_ws, size_t ws_size,
                              hipStream_t stream) {
    const int N = in_sizes[2];            // batch has N entries
    const int E = in_sizes[1] / 2;        // edge_index is (2,E)

    const float* x     = (const float*)d_in[0];
    const int*   ei    = (const int*)d_in[1];
    const int*   batch = (const int*)d_in[2];
    const float* W[4]  = {(const float*)d_in[3], (const float*)d_in[5],
                          (const float*)d_in[7], (const float*)d_in[9]};
    const float* B[4]  = {(const float*)d_in[4], (const float*)d_in[6],
                          (const float*)d_in[8], (const float*)d_in[10]};
    const float* G[3]  = {(const float*)d_in[11], (const float*)d_in[13], (const float*)d_in[15]};
    const float* BE[3] = {(const float*)d_in[12], (const float*)d_in[14], (const float*)d_in[16]};
    const float* w_lin = (const float*)d_in[17];
    const float* b_lin = (const float*)d_in[18];
    const int* src = ei;
    const int* dst = ei + E;

    // workspace carve-out
    char* wsp = (char*)d_ws;
    auto carve = [&](size_t bytes) { void* p = wsp; wsp += (bytes + 255) & ~(size_t)255; return p; };
    int*   cnt    = (int*)carve((size_t)N * 4);
    int*   rowptr = (int*)carve((size_t)(N + 1) * 4);
    int*   cursor = (int*)carve((size_t)N * 4);
    int*   csr    = (int*)carve((size_t)E * 4);
    float* invdeg = (float*)carve((size_t)N * 4);
    float* s1     = (float*)carve((size_t)N * 4);
    float* s2     = (float*)carve((size_t)N * 4);
    float* AGG    = (float*)carve((size_t)N * 384 * 4);       // 4*Fmax (Fmax=96)
    float* hA     = (float*)carve((size_t)N * 96 * 4);
    float* hB     = (float*)carve((size_t)N * 96 * 4);
    unsigned short* Wt = (unsigned short*)carve((size_t)96 * 1248 * 2);
    float* gs     = (float*)carve(2 * 96 * 4);
    float* zsum   = (float*)carve(NGRAPH * FOUT_LAST * 4);
    float* gcnt   = (float*)carve(NGRAPH * 4);

    // ---- CSR build (once, reused by all layers) ----
    hipMemsetAsync(cnt, 0, (size_t)N * 4, stream);
    hipMemsetAsync(cursor, 0, (size_t)N * 4, stream);
    degree_kernel<<<(E + 255) / 256, 256, 0, stream>>>(dst, cnt, E);
    scan_kernel<<<1, 1024, 0, stream>>>(cnt, rowptr, N);
    scatter_kernel<<<(E + 255) / 256, 256, 0, stream>>>(src, dst, rowptr, cursor, csr, E);
    node_scalars_kernel<<<(N + 255) / 256, 256, 0, stream>>>(cnt, invdeg, s1, s2, N);

    // ---- layers ----
    const int   Fin[4]  = {64, 96, 64, 32};
    const int   Fout[4] = {96, 64, 32, 20};
    const float* lin[4] = {x, hA, hB, hA};
    float*       lout[4] = {hA, hB, hA, hB};
    const int   use_bn[4]   = {1, 1, 0, 1};
    const int   use_relu[4] = {1, 1, 1, 0};
    const int   bn_idx[4]   = {0, 1, 0, 2};

    int mtiles = (N + 15) / 16;
    int gemmGrid = (mtiles + 3) / 4;
    int aggGrid = (N + 3) / 4;

    for (int L = 0; L < 4; ++L) {
        int F = Fin[L], Fo = Fout[L];
        int K13 = 13 * F;
        int Np = ((Fo + 15) / 16) * 16;

        // aggregation
        if (F == 32)       agg_kernel<1><<<aggGrid, 128, 0, stream>>>(lin[L], rowptr, csr, invdeg, AGG, N);
        else if (F == 64)  agg_kernel<2><<<aggGrid, 128, 0, stream>>>(lin[L], rowptr, csr, invdeg, AGG, N);
        else               agg_kernel<3><<<aggGrid, 128, 0, stream>>>(lin[L], rowptr, csr, invdeg, AGG, N);

        // weight prep to bf16 [Np, K13]
        int wtot = Np * K13;
        wprep_kernel<<<(wtot + 255) / 256, 256, 0, stream>>>(W[L], Wt, K13, Fo, wtot);

        // fused combine + GEMM (WMMA)
        if (L == 0)      pna_gemm_kernel<64, 96><<<gemmGrid, 128, 0, stream>>>(lin[L], AGG, s1, s2, Wt, B[L], lout[L], mtiles);
        else if (L == 1) pna_gemm_kernel<96, 64><<<gemmGrid, 128, 0, stream>>>(lin[L], AGG, s1, s2, Wt, B[L], lout[L], mtiles);
        else if (L == 2) pna_gemm_kernel<64, 32><<<gemmGrid, 128, 0, stream>>>(lin[L], AGG, s1, s2, Wt, B[L], lout[L], mtiles);
        else             pna_gemm_kernel<32, 20><<<gemmGrid, 128, 0, stream>>>(lin[L], AGG, s1, s2, Wt, B[L], lout[L], mtiles);

        // BN / ReLU
        int total = N * Fo;
        if (use_bn[L]) {
            hipMemsetAsync(gs, 0, (size_t)2 * Fo * 4, stream);
            bn_stats_kernel<<<512, 256, 0, stream>>>(lout[L], gs, total, Fo);
        }
        bn_apply_kernel<<<(total + 255) / 256, 256, 0, stream>>>(
            lout[L], gs, G[bn_idx[L]], BE[bn_idx[L]],
            total, Fo, 1.0f / (float)N, use_bn[L], use_relu[L]);
    }

    // ---- pool + head ----
    hipMemsetAsync(zsum, 0, NGRAPH * FOUT_LAST * 4, stream);
    hipMemsetAsync(gcnt, 0, NGRAPH * 4, stream);
    pool_cnt_kernel<<<(N + 255) / 256, 256, 0, stream>>>(batch, gcnt, N);
    pool_sum_kernel<<<(N * FOUT_LAST + 255) / 256, 256, 0, stream>>>(hB, batch, zsum, N);
    head_kernel<<<1, 64, 0, stream>>>(zsum, gcnt, w_lin, b_lin, (float*)d_out);
}
